// NoisyTopExpertsPerItemRouter_3719441678642
// MI455X (gfx1250) — matrix operand order repaired and checked by reading, not compiled, not run on hardware
//
#include <hip/hip_runtime.h>
#include <hip/hip_bf16.h>
#include <math.h>

typedef __attribute__((ext_vector_type(16))) _Float16 v16h;
typedef __attribute__((ext_vector_type(8)))  float    v8f;

// Problem constants
#define BB 8
#define SS 4096
#define DD 1024
#define EE 64
#define NOISE_STD (1.0f/64.0f)
#define INV_NOISE_STD 64.0f
#define NTOK (BB*SS)          // 32768
#define TOK_PER_BLK 64
#define LN_EPS_F 1e-5f

// Workspace layout (bytes)
#define WS_WH      0            // 64*1024 f16  = 131072 B
#define WS_C1      131072       // 64 f32
#define WS_C2      131328       // 64 f32
#define WS_IMP     131584       // 8*64 f32 (zeroed)
#define WS_GSUM    133632       // 64 f32 (zeroed)
#define WS_CNT     133888       // 64 f32 (zeroed)
#define WS_LSUMS   134144       // 2 f32 (zeroed)
#define WS_PACC    135168       // 4096*64 f32 = 1048576 B (zeroed)
#define WS_ZERO_BEG 131584
#define WS_ZERO_LEN (135168 - 131584 + SS*EE*4)

// ---------------------------------------------------------------------------
// Kernel 1: Wh[e,k] = f16(gamma[k]*W[e,k]);  c1[e]=sum(gamma*W); c2[e]=sum(beta*W)
// ---------------------------------------------------------------------------
__global__ __launch_bounds__(256) void prep_w_kernel(
    const float* __restrict__ W, const float* __restrict__ gamma,
    const float* __restrict__ beta, _Float16* __restrict__ Wh,
    float* __restrict__ c1, float* __restrict__ c2)
{
    const int e = blockIdx.x;
    const int tid = threadIdx.x;
    float s1 = 0.f, s2 = 0.f;
    for (int k = tid; k < DD; k += 256) {
        float w = W[e*DD + k];
        float wg = w * gamma[k];
        Wh[(size_t)e*DD + k] = (_Float16)wg;
        s1 += wg;
        s2 += w * beta[k];
    }
    __shared__ float a1[256], a2[256];
    a1[tid] = s1; a2[tid] = s2;
    __syncthreads();
    for (int o = 128; o > 0; o >>= 1) {
        if (tid < o) { a1[tid] += a1[tid+o]; a2[tid] += a2[tid+o]; }
        __syncthreads();
    }
    if (tid == 0) { c1[e] = a1[0]; c2[e] = a2[0]; }
}

// ---------------------------------------------------------------------------
// Kernel 2: main router. 512 blocks x 128 threads (4 waves x wave32).
// Wave w owns tokens [blk*64 + 16w, +16). WMMA 16x16x32 f16 -> f32.
// ---------------------------------------------------------------------------
__global__ __launch_bounds__(128) void router_main_kernel(
    const float* __restrict__ X, const float* __restrict__ noise,
    const _Float16* __restrict__ Wh,
    const float* __restrict__ c1, const float* __restrict__ c2,
    float* __restrict__ gates_out, float* __restrict__ imp_acc,
    float* __restrict__ gsum_acc, float* __restrict__ cnt_acc,
    float* __restrict__ p_acc)
{
    __shared__ float sh_logits[TOK_PER_BLK][EE + 2];
    __shared__ float sh_mu[TOK_PER_BLK], sh_rstd[TOK_PER_BLK];
    __shared__ float sh_c1[EE], sh_c2[EE];
    __shared__ float sh_imp[EE], sh_gsum[EE], sh_cnt[EE];

    const int tid  = threadIdx.x;
    const int wave = tid >> 5;
    const int lane = tid & 31;
    const int half = lane >> 4;     // lane group: 0 or 1
    const int m    = lane & 15;     // row/col within group
    const int tok0 = blockIdx.x * TOK_PER_BLK;
    const int wtok = tok0 + wave * 16;

    if (tid < EE)       { sh_c1[tid] = c1[tid]; sh_imp[tid] = 0.f; sh_gsum[tid] = 0.f; sh_cnt[tid] = 0.f; }
    else if (tid < 128) { sh_c2[tid - EE] = c2[tid - EE]; }

    // ---- Phase A: LayerNorm statistics (one row reduced by whole wave) ----
    for (int t = 0; t < 16; ++t) {
        const float4* row = (const float4*)(X + (size_t)(wtok + t) * DD);
        float s = 0.f, q = 0.f;
#pragma unroll
        for (int i = 0; i < 8; ++i) {
            float4 v = row[lane + 32*i];
            s += v.x + v.y + v.z + v.w;
            q += v.x*v.x + v.y*v.y + v.z*v.z + v.w*v.w;
        }
#pragma unroll
        for (int o = 16; o >= 1; o >>= 1) {
            s += __shfl_xor(s, o, 32);
            q += __shfl_xor(q, o, 32);
        }
        if (lane == 0) {
            float mu  = s * (1.f/(float)DD);
            float var = q * (1.f/(float)DD) - mu*mu;
            sh_mu[wave*16 + t]   = mu;
            sh_rstd[wave*16 + t] = rsqrtf(var + LN_EPS_F);
        }
    }
    __syncthreads();

    // ---- Phase B: GEMM via v_wmma_f32_16x16x32_f16 ----
    v8f acc[4];
#pragma unroll
    for (int n = 0; n < 4; ++n)
        acc[n] = (v8f){0.f,0.f,0.f,0.f,0.f,0.f,0.f,0.f};

    const float* Arow = X + (size_t)(wtok + m) * DD;   // this lane's token row
    for (int kt = 0; kt < 32; ++kt) {
        const int kb = kt * 32;
        // A fragment: halves 0..15 = K[kb+16h .. +16)
        const float4* pa0 = (const float4*)(Arow + kb + 16*half);
        const float4* pa1 = (const float4*)(Arow + kb + 16*half + 8);
        float4 f0 = pa0[0], f1 = pa0[1];
        float4 f2 = pa1[0], f3 = pa1[1];
        __builtin_prefetch(Arow + kb + 256, 0, 1);     // global_prefetch next K tiles
        v16h a;
        a[0]=(_Float16)f0.x;  a[1]=(_Float16)f0.y;  a[2]=(_Float16)f0.z;  a[3]=(_Float16)f0.w;
        a[4]=(_Float16)f1.x;  a[5]=(_Float16)f1.y;  a[6]=(_Float16)f1.z;  a[7]=(_Float16)f1.w;
        a[8]=(_Float16)f2.x;  a[9]=(_Float16)f2.y;  a[10]=(_Float16)f2.z; a[11]=(_Float16)f2.w;
        a[12]=(_Float16)f3.x; a[13]=(_Float16)f3.y; a[14]=(_Float16)f3.z; a[15]=(_Float16)f3.w;
#pragma unroll
        for (int n = 0; n < 4; ++n) {
            // B fragment: lane holds 16 contiguous halves of expert row e at K = kb + 16*half
            const v16h b = *reinterpret_cast<const v16h*>(
                Wh + (size_t)(n*16 + m) * DD + kb + 16*half);
            acc[n] = __builtin_amdgcn_wmma_f32_16x16x32_f16(
                false, a, false, b, (short)0, acc[n], false, false);
        }
    }

    // C layout: VGPR r -> row (r + 8*half), col = m (+16n)
#pragma unroll
    for (int n = 0; n < 4; ++n)
#pragma unroll
        for (int r = 0; r < 8; ++r)
            sh_logits[wave*16 + r + 8*half][n*16 + m] = acc[n][r];
    __syncthreads();

    // ---- Phase C: softmax / losses. Two lanes per token (32 experts each) ----
    const int tl  = tid >> 1;        // local token 0..63
    const int par = tid & 1;         // expert half
    const int tg  = tok0 + tl;       // global token = b*S + s
    const int bidx = tg >> 12;       // /4096
    const int sidx = tg & 4095;
    const float mu = sh_mu[tl], rstd = sh_rstd[tl];
    const float mur = mu * rstd;

    float l[32], ln[32];
    float mx = -1e30f;
#pragma unroll
    for (int j = 0; j < 32; ++j) {
        const int e = par*32 + j;
        float v = rstd * sh_logits[tl][e] - mur * sh_c1[e] + sh_c2[e];
        l[j] = v;
        mx = fmaxf(mx, v);
    }
    mx = fmaxf(mx, __shfl_xor(mx, 1, 32));            // row max over 64

    float se = 0.f;
#pragma unroll
    for (int j = 0; j < 32; ++j) { l[j] -= mx; se += __expf(l[j]); }
    se += __shfl_xor(se, 1, 32);
    const float inv = 1.f / se;

    // noisy logits
    const float* nrow = noise + (size_t)tg * EE + par*32;
    float mx2 = -1e30f;
#pragma unroll
    for (int j = 0; j < 32; ++j) {
        ln[j] = l[j] + NOISE_STD * nrow[j];
        mx2 = fmaxf(mx2, ln[j]);
    }
    mx2 = fmaxf(mx2, __shfl_xor(mx2, 1, 32));
    float sn = 0.f;
#pragma unroll
    for (int j = 0; j < 32; ++j) sn += __expf(ln[j] - mx2);
    sn += __shfl_xor(sn, 1, 32);
    const float invn = 1.f / sn;

    // top-2 of noisy logits (threshold = 2nd largest) and argmax
    float t1 = -1e30f, t2 = -1e30f; int a1 = par*32;
#pragma unroll
    for (int j = 0; j < 32; ++j) {
        float v = ln[j];
        if (v > t1) { t2 = t1; t1 = v; a1 = par*32 + j; }
        else        { t2 = fmaxf(t2, v); }
    }
    const float o1 = __shfl_xor(t1, 1, 32);
    const float o2 = __shfl_xor(t2, 1, 32);
    const int   oa = __shfl_xor(a1, 1, 32);
    const float thr = (t1 > o1) ? fmaxf(t2, o1) : fmaxf(o2, t1);
    // combined argmax; tie -> lower expert index (par0 side)
    int widx; 
    if (par == 0) widx = (o1 > t1) ? oa : a1;
    else          widx = (t1 > o1) ? a1 : oa;

    float* orow = gates_out + (size_t)tg * EE + par*32;
    float* prow = p_acc + (size_t)sidx * EE + par*32;
#pragma unroll
    for (int j = 0; j < 32; ++j) {
        const int e = par*32 + j;
        float gate = __expf(l[j]) * inv;                 // clean softmax
        atomicAdd(&sh_imp[e], gate);
        float gn = __expf(ln[j] - mx2) * invn;           // noisy softmax
        orow[j] = gn;
        atomicAdd(&sh_gsum[e], gn);
        float z = (thr - l[j]) * INV_NOISE_STD;
        z = fminf(fmaxf(z, -10.f), 10.f);
        float p = 0.5f * (1.0f + erff(z * 0.70710678118f));
        atomicAdd(&prow[j], p);                          // sum over batch axis
    }
    if (par == 0) atomicAdd(&sh_cnt[widx], 1.0f);
    __syncthreads();

    if (tid < EE) {
        atomicAdd(&imp_acc[bidx*EE + tid], sh_imp[tid]); // all tokens in blk share b
        atomicAdd(&gsum_acc[tid], sh_gsum[tid]);
        atomicAdd(&cnt_acc[tid], sh_cnt[tid]);
    }
}

// ---------------------------------------------------------------------------
// Kernel 3: reduce p_acc (S*E) -> sum(p_mean), sum(p_mean^2)
// ---------------------------------------------------------------------------
__global__ __launch_bounds__(256) void reduce_p_kernel(
    const float* __restrict__ p_acc, float* __restrict__ lsums)
{
    __shared__ float r1[256], r2[256];
    const int tid = threadIdx.x;
    float s1 = 0.f, s2 = 0.f;
    for (int i = blockIdx.x * 256 + tid; i < SS*EE; i += gridDim.x * 256) {
        float pm = p_acc[i] * (1.f/(float)BB);
        s1 += pm; s2 += pm * pm;
    }
    r1[tid] = s1; r2[tid] = s2;
    __syncthreads();
    for (int o = 128; o > 0; o >>= 1) {
        if (tid < o) { r1[tid] += r1[tid+o]; r2[tid] += r2[tid+o]; }
        __syncthreads();
    }
    if (tid == 0) { atomicAdd(&lsums[0], r1[0]); atomicAdd(&lsums[1], r2[0]); }
}

// ---------------------------------------------------------------------------
// Kernel 4: finalize the 4 scalars
// ---------------------------------------------------------------------------
__global__ void finalize_kernel(
    const float* __restrict__ imp_acc, const float* __restrict__ gsum,
    const float* __restrict__ cnt, const float* __restrict__ lsums,
    float* __restrict__ out_scalars)
{
    if (threadIdx.x != 0 || blockIdx.x != 0) return;
    // importance: mean over b of (std(imp,ddof=1)/mean)^2
    float imp_loss = 0.f;
    for (int b = 0; b < BB; ++b) {
        const float* v = imp_acc + b*EE;
        float s = 0.f;
        for (int e = 0; e < EE; ++e) s += v[e];
        float mean = s / (float)EE;
        float var = 0.f;
        for (int e = 0; e < EE; ++e) { float d = v[e] - mean; var += d*d; }
        var /= (float)(EE - 1);
        imp_loss += var / (mean * mean);
    }
    imp_loss /= (float)BB;
    // gshard: mean(mean_top1 * mean_gates) * E^2
    float gs = 0.f;
    for (int e = 0; e < EE; ++e)
        gs += (cnt[e] / (float)NTOK) * (gsum[e] / (float)NTOK);
    gs = (gs / (float)EE) * (float)EE * (float)EE;
    // load: cv^2 (ddof=0) of p_mean over S*E
    const float N = (float)(SS * EE);
    float mean = lsums[0] / N;
    float var  = lsums[1] / N - mean * mean;
    float load = var / (mean * mean);
    float aux = 0.0f * gs + 1.0f * imp_loss + 1.0f * load;
    out_scalars[0] = aux;
    out_scalars[1] = gs;
    out_scalars[2] = imp_loss;
    out_scalars[3] = load;
}

// ---------------------------------------------------------------------------
extern "C" void kernel_launch(void* const* d_in, const int* in_sizes, int n_in,
                              void* d_out, int out_size, void* d_ws, size_t ws_size,
                              hipStream_t stream) {
    (void)in_sizes; (void)n_in; (void)out_size; (void)ws_size;
    const float* X     = (const float*)d_in[0];  // inputs  (B,S,D)
    const float* noise = (const float*)d_in[1];  // noise   (B,S,E)
    const float* gamma = (const float*)d_in[2];  // ln_gamma(D)
    const float* beta  = (const float*)d_in[3];  // ln_beta (D)
    const float* W     = (const float*)d_in[4];  // W       (E,D)
    float* out = (float*)d_out;                  // gates (B*S*E) + 4 scalars

    char* ws = (char*)d_ws;
    _Float16* Wh  = (_Float16*)(ws + WS_WH);
    float* c1     = (float*)(ws + WS_C1);
    float* c2     = (float*)(ws + WS_C2);
    float* imp    = (float*)(ws + WS_IMP);
    float* gsum   = (float*)(ws + WS_GSUM);
    float* cnt    = (float*)(ws + WS_CNT);
    float* lsums  = (float*)(ws + WS_LSUMS);
    float* p_acc  = (float*)(ws + WS_PACC);

    // zero all cross-block accumulators every call (graph-replay determinism)
    hipMemsetAsync(ws + WS_ZERO_BEG, 0, WS_ZERO_LEN, stream);

    prep_w_kernel<<<EE, 256, 0, stream>>>(W, gamma, beta, Wh, c1, c2);
    router_main_kernel<<<NTOK / TOK_PER_BLK, 128, 0, stream>>>(
        X, noise, Wh, c1, c2, out, imp, gsum, cnt, p_acc);
    reduce_p_kernel<<<128, 256, 0, stream>>>(p_acc, lsums);
    finalize_kernel<<<1, 64, 0, stream>>>(imp, gsum, cnt, lsums, out + (size_t)NTOK * EE);
}
